// EncoderLTC_20186346291692
// MI455X (gfx1250) — matrix operand order repaired and checked by reading, not compiled, not run on hardware
//
#include <hip/hip_runtime.h>
#include <hip/hip_bf16.h>
#include <math.h>

// Bidirectional LTC encoder for MI455X (gfx1250, wave32, WMMA).
// grid=2 (fw/bw), block=1024 (32 waves on one WGP, 8 waves/SIMD).
// Wave w owns output columns [16w,16w+16); the f32 hidden state lives in WMMA
// accumulator fragments across all 512 steps; h is re-shared per unfold via a
// padded bf16 LDS tile (conflict-free ds_load_b128 A-fragments). W_rec/W_in
// stream from L2 as B-fragments each unfold -- an opaque *integer offset*
// blocks LICM from hoisting (and scratch-spilling) the loop-invariant weight
// loads while preserving pointer provenance, so they lower to global_load
// (saddr form, LOADcnt only) instead of flat_load (which would pollute DScnt
// and falsely serialize against the LDS A-fragment pipeline).
// The per-step input projection `pre` is thread-private (C-fragment layout)
// and parked in per-thread LDS. The next step's embedding gather is issued
// before the unfolds and committed to LDS after them, hiding its L2/HBM
// latency behind ~400 WMMAs of recurrent compute.

typedef __bf16 bf16_t;
typedef __attribute__((ext_vector_type(16))) __bf16 v16bf;
typedef __attribute__((ext_vector_type(8)))  __bf16 v8bf;
typedef __attribute__((ext_vector_type(8)))  float  v8f;

#define S_LEN  512
#define BATCH  64
#define EMB    256
#define HID    512
#define NUNF   6
#define LTC_DT (1.0f/6.0f)

#define HP (HID + 8)   // padded LDS stride for h (bf16 elems)
#define EP (EMB + 8)   // padded LDS stride for e (bf16 elems)
#define PS 36          // per-thread pre stride (dwords): 32 data + 4 pad

__device__ __forceinline__ v8f wmma_bf16(v16bf a, v16bf b, v8f c) {
  // D = A(16x32) * B(32x16) + C, f32 accumulate
  return __builtin_amdgcn_wmma_f32_16x16x32_bf16(false, a, false, b,
                                                 (short)0, c, false, false);
}

__device__ __forceinline__ v8f zero8() {
  v8f z;
#pragma unroll
  for (int i = 0; i < 8; ++i) z[i] = 0.0f;
  return z;
}

// Opaque zero offset: redefined by empty asm each use, so weight loads based
// on (W + opaque_zero()) cannot be hoisted out of the recurrence loops (which
// would spill B-fragments to scratch). Unlike laundering the pointer itself,
// this preserves address-space provenance -> global_load, not flat_load.
__device__ __forceinline__ int opaque_zero() {
  int z = 0;
  asm volatile("" : "+s"(z));
  return z;
}

// A fragment (16x32 bf16, M x K) from row-major [.,ld] source.
// Lane L: M = mbase + L%16; elems 0..7 -> K = kt*32 + (L/16)*8 + i,
//         elems 8..15 -> K = kt*32 + 16 + (L/16)*8 + (i-8).
__device__ __forceinline__ v16bf load_afrag(const bf16_t* __restrict__ base, int ld,
                                            int mbase, int ktile, int lane) {
  const int m  = mbase + (lane & 15);
  const int kh = (lane >> 4) << 3;
  const bf16_t* p = base + m * ld + ktile * 32 + kh;
  v8bf lo = *(const v8bf*)(p);
  v8bf hi = *(const v8bf*)(p + 16);
  v16bf r;
#pragma unroll
  for (int i = 0; i < 8; ++i) { r[i] = lo[i]; r[8 + i] = hi[i]; }
  return r;
}

// B fragment (32x16 bf16, K x N) == W^T tile from row-major W[n][k].
// Lane L: N = nbase + L%16; elems 0..7 -> K = kt*32 + (L/16)*8 + i,
//         elems 8..15 -> K = kt*32 + 16 + (L/16)*8 + (i-8).
__device__ __forceinline__ v16bf load_bfrag(const bf16_t* __restrict__ base, int ld,
                                            int nbase, int ktile, int lane) {
  const int n  = nbase + (lane & 15);
  const int kb = ktile * 32 + ((lane >> 4) << 3);
  const bf16_t* p = base + n * ld + kb;
  v8bf lo = *(const v8bf*)(p);
  v8bf hi = *(const v8bf*)(p + 16);
  v16bf r;
#pragma unroll
  for (int i = 0; i < 8; ++i) { r[i] = lo[i]; r[8 + i] = hi[i]; }
  return r;
}

struct GatherRegs { float4 a0, a1, a2, a3; };

__device__ __forceinline__ void gather_issue(const int* __restrict__ x,
                                             const float* __restrict__ emb,
                                             int src, int tid, GatherRegs& g) {
  const int b = tid >> 4;
  const int c = (tid & 15) << 4;
  const int row = x[src * BATCH + b];
  const float4* er = (const float4*)(emb + (long)row * EMB + c);
  g.a0 = er[0]; g.a1 = er[1]; g.a2 = er[2]; g.a3 = er[3];
}

__device__ __forceinline__ void gather_commit(bf16_t* e_sh, int tid,
                                              const GatherRegs& g) {
  const int b = tid >> 4;
  const int c = (tid & 15) << 4;
  bf16_t* dst = e_sh + b * EP + c;
  v8bf o0, o1;
  o0[0]=(bf16_t)g.a0.x; o0[1]=(bf16_t)g.a0.y; o0[2]=(bf16_t)g.a0.z; o0[3]=(bf16_t)g.a0.w;
  o0[4]=(bf16_t)g.a1.x; o0[5]=(bf16_t)g.a1.y; o0[6]=(bf16_t)g.a1.z; o0[7]=(bf16_t)g.a1.w;
  o1[0]=(bf16_t)g.a2.x; o1[1]=(bf16_t)g.a2.y; o1[2]=(bf16_t)g.a2.z; o1[3]=(bf16_t)g.a2.w;
  o1[4]=(bf16_t)g.a3.x; o1[5]=(bf16_t)g.a3.y; o1[6]=(bf16_t)g.a3.z; o1[7]=(bf16_t)g.a3.w;
  *(v8bf*)(dst)     = o0;
  *(v8bf*)(dst + 8) = o1;
}

__global__ void cvt_bf16_kernel(const float* __restrict__ s,
                                bf16_t* __restrict__ d, int n) {
  int i = blockIdx.x * blockDim.x + threadIdx.x;
  if (i < n) d[i] = (bf16_t)s[i];
}

__global__ void __launch_bounds__(1024, 1)
ltc_scan_kernel(const int* __restrict__ x, const float* __restrict__ emb,
                const bf16_t* __restrict__ wi_fw, const bf16_t* __restrict__ wr_fw,
                const float* __restrict__ b_fw,  const float* __restrict__ tau_fw,
                const bf16_t* __restrict__ wi_bw, const bf16_t* __restrict__ wr_bw,
                const float* __restrict__ b_bw,  const float* __restrict__ tau_bw,
                float* __restrict__ enc_out, bf16_t* __restrict__ hfin_bf)
{
  const int dir  = blockIdx.x;            // 0 = forward, 1 = backward
  const int tid  = threadIdx.x;
  const int wv   = tid >> 5;              // wave id == N-tile id (0..31)
  const int lane = tid & 31;
  const int nloc = (wv << 4) + (lane & 15);  // this lane's output column

  const bf16_t* Wi = dir ? wi_bw  : wi_fw;
  const bf16_t* Wr = dir ? wr_bw  : wr_fw;
  const float*  bb = dir ? b_bw   : b_fw;
  const float*  tt = dir ? tau_bw : tau_fw;

  const float bias = bb[nloc];
  const float itau = 1.0f / tt[nloc];

  // LDS: [ pre (f32, thread-private) | h (bf16, shared) | e (bf16, shared) ]
  extern __shared__ float smem_f[];
  float*  pre_my = smem_f + tid * PS;                       // 32 floats/thread
  bf16_t* h_sh   = (bf16_t*)(smem_f + 1024 * PS);           // [BATCH][HP]
  bf16_t* e_sh   = h_sh + BATCH * HP;                       // [BATCH][EP]

  v8f h[4];                               // persistent f32 state fragments
#pragma unroll
  for (int m = 0; m < 4; ++m) h[m] = zero8();

  for (int idx = tid; idx < BATCH * HP; idx += 1024) h_sh[idx] = (bf16_t)0.0f;

  // prologue: gather e for the first step
  {
    GatherRegs g;
    gather_issue(x, emb, dir ? (S_LEN - 1) : 0, tid, g);
    gather_commit(e_sh, tid, g);
  }
  __syncthreads();

  for (int t = 0; t < S_LEN; ++t) {
    const int src = dir ? (S_LEN - 1 - t) : t;

    // ---- pre = e @ W_in^T (this wave's 16-column slice), K = 256,
    //      then park thread-private result in LDS (no barrier needed).
    {
      const bf16_t* WiT = Wi + opaque_zero();   // block hoist across t
      v8f acc[4];
#pragma unroll
      for (int m = 0; m < 4; ++m) acc[m] = zero8();
      for (int kt = 0; kt < EMB / 32; ++kt) {
        v16bf bf = load_bfrag(WiT, EMB, wv << 4, kt, lane);
#pragma unroll
        for (int m = 0; m < 4; ++m) {
          v16bf af = load_afrag(e_sh, EP, m << 4, kt, lane);
          acc[m] = wmma_bf16(af, bf, acc[m]);
        }
      }
#pragma unroll
      for (int m = 0; m < 4; ++m)
#pragma unroll
        for (int i = 0; i < 8; ++i) pre_my[m * 8 + i] = acc[m][i];
    }
    __syncthreads();   // all waves finished reading e_sh

    // ---- issue next step's gather; loads retire during the unfolds
    const bool more = (t + 1 < S_LEN);
    GatherRegs g;
    if (more) gather_issue(x, emb, dir ? (S_LEN - 2 - t) : (t + 1), tid, g);

    // ---- 6 ODE unfolds: f = tanh(pre + h@Wrec^T + b); h += DT*(f - h/tau)
    for (int u = 0; u < NUNF; ++u) {
      const bf16_t* WrU = Wr + opaque_zero();   // block hoist across u
      v8f acc[4];
#pragma unroll
      for (int m = 0; m < 4; ++m) acc[m] = zero8();
      for (int kt = 0; kt < HID / 32; ++kt) {
        v16bf bf = load_bfrag(WrU, HID, wv << 4, kt, lane);  // L2-resident
#pragma unroll
        for (int m = 0; m < 4; ++m) {
          v16bf af = load_afrag(h_sh, HP, m << 4, kt, lane); // ds_load_b128
          acc[m] = wmma_bf16(af, bf, acc[m]);
        }
      }
      __syncthreads();   // all waves finished reading h_sh
#pragma unroll
      for (int m = 0; m < 4; ++m) {
#pragma unroll
        for (int i = 0; i < 8; ++i) {
          float f  = tanhf(pre_my[m * 8 + i] + acc[m][i] + bias);
          float hv = h[m][i];
          hv += LTC_DT * (f - hv * itau);
          h[m][i] = hv;
          const int mr = (m << 4) + ((lane >> 4) << 3) + i;  // C-layout row
          h_sh[mr * HP + nloc] = (bf16_t)hv;
        }
      }
      __syncthreads();   // new h visible to all waves
    }

    // ---- commit next step's embeddings (loads long since in flight)
    if (more) gather_commit(e_sh, tid, g);

    // ---- encoder output slice (f32), at original sequence position `src`
    {
      float* op = enc_out + ((long)src * BATCH) * (2 * HID) + dir * HID + nloc;
#pragma unroll
      for (int m = 0; m < 4; ++m)
#pragma unroll
        for (int i = 0; i < 8; ++i) {
          const int mr = (m << 4) + ((lane >> 4) << 3) + i;
          op[(long)mr * (2 * HID)] = h[m][i];
        }
    }
    __syncthreads();   // e_sh commit visible before next pre reads it
  }

  // ---- final hidden (bf16) -> workspace for FC kernel, concat layout [64][1024]
  {
    bf16_t* hp = hfin_bf + dir * HID + nloc;
#pragma unroll
    for (int m = 0; m < 4; ++m)
#pragma unroll
      for (int i = 0; i < 8; ++i) {
        const int mr = (m << 4) + ((lane >> 4) << 3) + i;
        hp[mr * (2 * HID)] = (bf16_t)h[m][i];
      }
  }
}

__global__ void __launch_bounds__(1024, 1)
fc_kernel(const bf16_t* __restrict__ hfin, const bf16_t* __restrict__ fcw,
          const float* __restrict__ fcb, float* __restrict__ dec_out)
{
  const int tid = threadIdx.x, wv = tid >> 5, lane = tid & 31;
  const int nloc = (wv << 4) + (lane & 15);
  v8f acc[4];
#pragma unroll
  for (int m = 0; m < 4; ++m) acc[m] = zero8();
  for (int kt = 0; kt < (2 * HID) / 32; ++kt) {
    v16bf bf = load_bfrag(fcw, 2 * HID, wv << 4, kt, lane);
#pragma unroll
    for (int m = 0; m < 4; ++m) {
      v16bf af = load_afrag(hfin, 2 * HID, m << 4, kt, lane);
      acc[m] = wmma_bf16(af, bf, acc[m]);
    }
  }
  const float bias = fcb[nloc];
#pragma unroll
  for (int m = 0; m < 4; ++m)
#pragma unroll
    for (int i = 0; i < 8; ++i) {
      const int mr = (m << 4) + ((lane >> 4) << 3) + i;
      dec_out[mr * HID + nloc] = tanhf(acc[m][i] + bias);
    }
}

extern "C" void kernel_launch(void* const* d_in, const int* in_sizes, int n_in,
                              void* d_out, int out_size, void* d_ws, size_t ws_size,
                              hipStream_t stream) {
  const int*   x      = (const int*)  d_in[0];
  const float* emb    = (const float*)d_in[1];
  const float* wi_fw  = (const float*)d_in[2];
  const float* wr_fw  = (const float*)d_in[3];
  const float* b_fw   = (const float*)d_in[4];
  const float* tau_fw = (const float*)d_in[5];
  const float* wi_bw  = (const float*)d_in[6];
  const float* wr_bw  = (const float*)d_in[7];
  const float* b_bw   = (const float*)d_in[8];
  const float* tau_bw = (const float*)d_in[9];
  const float* fc_w   = (const float*)d_in[10];
  const float* fc_b   = (const float*)d_in[11];

  // workspace: bf16 weight copies (<3 MB) + final hidden staging
  bf16_t* ws = (bf16_t*)d_ws;
  const int WI_N = HID * EMB;          // 131072
  const int WR_N = HID * HID;          // 262144
  const int FC_N = HID * 2 * HID;      // 524288
  bf16_t* wi_fw_b = ws;
  bf16_t* wr_fw_b = wi_fw_b + WI_N;
  bf16_t* wi_bw_b = wr_fw_b + WR_N;
  bf16_t* wr_bw_b = wi_bw_b + WI_N;
  bf16_t* fc_w_b  = wr_bw_b + WR_N;
  bf16_t* hfin    = fc_w_b  + FC_N;    // [64][1024] bf16

  cvt_bf16_kernel<<<(WI_N + 255) / 256, 256, 0, stream>>>(wi_fw, wi_fw_b, WI_N);
  cvt_bf16_kernel<<<(WR_N + 255) / 256, 256, 0, stream>>>(wr_fw, wr_fw_b, WR_N);
  cvt_bf16_kernel<<<(WI_N + 255) / 256, 256, 0, stream>>>(wi_bw, wi_bw_b, WI_N);
  cvt_bf16_kernel<<<(WR_N + 255) / 256, 256, 0, stream>>>(wr_bw, wr_bw_b, WR_N);
  cvt_bf16_kernel<<<(FC_N + 255) / 256, 256, 0, stream>>>(fc_w,  fc_w_b,  FC_N);

  // LDS: pre (1024*36 dwords = 144 KB) + h (66.5 KB) + e (33.8 KB) ~= 242 KB
  const size_t smem = (size_t)1024 * PS * sizeof(float)
                    + (size_t)(BATCH * HP + BATCH * EP) * sizeof(bf16_t);
  ltc_scan_kernel<<<2, 1024, smem, stream>>>(x, emb,
                                             wi_fw_b, wr_fw_b, b_fw, tau_fw,
                                             wi_bw_b, wr_bw_b, b_bw, tau_bw,
                                             (float*)d_out, hfin);

  float* dec = (float*)d_out + (size_t)S_LEN * BATCH * 2 * HID;
  fc_kernel<<<1, 1024, 0, stream>>>(hfin, fc_w_b, fc_b, dec);
}
